// LorentzTransformer_70085276336566
// MI455X (gfx1250) — matrix operand order repaired and checked by reading, not compiled, not run on hardware
//
#include <hip/hip_runtime.h>
#include <hip/hip_bf16.h>

typedef __bf16 bf16_t;
typedef __attribute__((ext_vector_type(16))) __bf16 v16bf;
typedef __attribute__((ext_vector_type(8)))  float  v8f;
typedef __attribute__((ext_vector_type(4)))  unsigned u32x4;

#define ALPHA_L 0.25f
#define Bsz 2
#define Lseq 1024
#define Dm 1024
#define Hh 16
#define DHd 64
#define MLrows (Bsz * Lseq) /* 2048 */

static __device__ __forceinline__ bf16_t f2bf(float f) {
  unsigned u = __builtin_bit_cast(unsigned, f);
  u += 0x7fffu + ((u >> 16) & 1u);           // round-to-nearest-even
  unsigned short h = (unsigned short)(u >> 16);
  return __builtin_bit_cast(bf16_t, h);
}

static __device__ __forceinline__ v8f zero8() {
  v8f z;
#pragma unroll
  for (int i = 0; i < 8; ++i) z[i] = 0.f;
  return z;
}

// A-fragment (16x32, 16-bit): lane l -> row (l&15); lanes 0-15 hold K {k0..k0+7, k0+16..k0+23},
// lanes 16-31 hold K {k0+8..k0+15, k0+24..k0+31}.  X row-major bf16, row stride ld.
static __device__ __forceinline__ v16bf load_a(const bf16_t* X, int row0, int ld, int k0, int lane) {
  int hi = lane >> 4;
  const bf16_t* p = X + (size_t)(row0 + (lane & 15)) * ld + k0 + hi * 8;
  union { v16bf v; u32x4 q[2]; } u;
  u.q[0] = *(const u32x4*)p;
  u.q[1] = *(const u32x4*)(p + 16);
  return u.v;
}

// B-fragment (32x16, 16-bit) with B[k,n] = X[n][k], X row-major:
// lane l -> column (l&15); lanes 0-15 hold K {k0..k0+7, k0+16..k0+23}, lanes 16-31 hold K {k0+8..k0+15, k0+24..k0+31}.
static __device__ __forceinline__ v16bf load_b(const bf16_t* X, int row0, int ld, int k0, int lane) {
  const bf16_t* p = X + (size_t)(row0 + (lane & 15)) * ld + k0 + (lane >> 4) * 8;
  union { v16bf v; u32x4 q[2]; } u;
  u.q[0] = *(const u32x4*)p;
  u.q[1] = *(const u32x4*)(p + 16);
  return u.v;
}

static __device__ __forceinline__ v8f wmma_bf(v16bf a, v16bf b, v8f c) {
  return __builtin_amdgcn_wmma_f32_16x16x32_bf16(false, a, false, b, (short)0, c, false, false);
}

// ---------------- f32 -> bf16 conversion ----------------
__global__ void cvt_kernel(const float* __restrict__ in, bf16_t* __restrict__ out, int n) {
  int i = blockIdx.x * blockDim.x + threadIdx.x;
  if (i < n) out[i] = f2bf(in[i]);
}

// ---------------- fused QKV GEMM ----------------
// one wave = 16x64 output tile; Q fp32 (B,H,L,DH); K bf16 (B,H,L,DH); V bf16 transposed (B,H,DH,L)
__global__ void __launch_bounds__(256) gemm_qkv_kernel(
    const bf16_t* __restrict__ xbf, const bf16_t* __restrict__ wq,
    const bf16_t* __restrict__ wk,  const bf16_t* __restrict__ wv,
    float* __restrict__ Qf, bf16_t* __restrict__ Kb, bf16_t* __restrict__ Vt) {
  int lane = threadIdx.x & 31;
  int t = blockIdx.x * 8 + (threadIdx.x >> 5);  // 2048 waves total
  int m0 = (t >> 4) * 16;                       // 128 row tiles
  int n0 = (t & 15) * 64;                       // 16 col tiles of 64
  v8f acc[3][4];
#pragma unroll
  for (int s = 0; s < 3; ++s)
#pragma unroll
    for (int j = 0; j < 4; ++j) acc[s][j] = zero8();

  for (int k0 = 0; k0 < Dm; k0 += 32) {
    v16bf a = load_a(xbf, m0, Dm, k0, lane);
#pragma unroll
    for (int j = 0; j < 4; ++j) {
      v16bf bq = load_b(wq, n0 + j * 16, Dm, k0, lane);
      acc[0][j] = wmma_bf(a, bq, acc[0][j]);
      v16bf bk = load_b(wk, n0 + j * 16, Dm, k0, lane);
      acc[1][j] = wmma_bf(a, bk, acc[1][j]);
      v16bf bv = load_b(wv, n0 + j * 16, Dm, k0, lane);
      acc[2][j] = wmma_bf(a, bv, acc[2][j]);
    }
  }
  int n = lane & 15, mAdd = (lane >> 4) * 8;   // C frag: m = r + 8*(lane>=16), n = lane&15
#pragma unroll
  for (int j = 0; j < 4; ++j)
#pragma unroll
    for (int r = 0; r < 8; ++r) {
      int row = m0 + mAdd + r;
      int b = row >> 10, l = row & 1023;
      int col = n0 + j * 16 + n;
      int h = col >> 6, d = col & 63;
      size_t qi = (((size_t)(b * Hh + h)) * Lseq + l) * DHd + d;
      Qf[qi] = acc[0][j][r];
      Kb[qi] = f2bf(acc[1][j][r]);
      size_t vi = (((size_t)(b * Hh + h)) * DHd + d) * Lseq + l;
      Vt[vi] = f2bf(acc[2][j][r]);
    }
}

// ---------------- Lorentz correction: Qeff = Q * (1 - 2*alpha*sf*m) ----------------
__global__ void __launch_bounds__(256) lorentz_kernel(const float* __restrict__ Qf,
    const unsigned char* __restrict__ tmask, bf16_t* __restrict__ Qe) {
  int lane = threadIdx.x & 31;
  int wid = blockIdx.x * 8 + (threadIdx.x >> 5);  // 32768 rows (b,h,l)
  int h = (wid >> 10) & 15;
  const float* q = Qf + (size_t)wid * DHd;
  float q0 = q[lane], q1 = q[lane + 32];
  float m0 = tmask[h * DHd + lane] ? 1.f : 0.f;
  float m1 = tmask[h * DHd + lane + 32] ? 1.f : 0.f;
  float ss = q0 * q0 + q1 * q1;
  float st = m0 * q0 * q0 + m1 * q1 * q1;
#pragma unroll
  for (int off = 16; off > 0; off >>= 1) {
    ss += __shfl_xor(ss, off);
    st += __shfl_xor(st, off);
  }
  float qn = sqrtf(ss), qtn = sqrtf(st);
  float sf = (qtn > 1e-6f) ? (qn / fmaxf(qtn, 1e-8f)) : 0.f;
  float c = 1.f - 2.f * ALPHA_L * sf;
  bf16_t* o = Qe + (size_t)wid * DHd;
  o[lane]      = f2bf(m0 != 0.f ? q0 * c : q0);
  o[lane + 32] = f2bf(m1 != 0.f ? q1 * c : q1);
}

// ---------------- causal flash attention, one wave per 16-query tile ----------------
__global__ void __launch_bounds__(256) flash_kernel(const bf16_t* __restrict__ Qe,
    const bf16_t* __restrict__ Kb, const bf16_t* __restrict__ Vt, bf16_t* __restrict__ Abf) {
  __shared__ __attribute__((aligned(16))) bf16_t lds[8][16 * 32];  // per-wave P tile
  int lane = threadIdx.x & 31;
  int wvid = threadIdx.x >> 5;
  int gw = blockIdx.x * 8 + wvid;     // 2048 waves
  int qt = gw & 63;
  int bh = gw >> 6;                   // b*16 + h
  int q0 = qt * 16;
  const bf16_t* Qp = Qe + (size_t)bh * Lseq * DHd;
  const bf16_t* Kp = Kb + (size_t)bh * Lseq * DHd;
  const bf16_t* Vp = Vt + (size_t)bh * DHd * Lseq;

  v16bf a0 = load_a(Qp, q0, DHd, 0, lane);
  v16bf a1 = load_a(Qp, q0, DHd, 32, lane);
  float mrun[8], lrun[8];
  v8f O[4];
#pragma unroll
  for (int r = 0; r < 8; ++r) { mrun[r] = -3.0e38f; lrun[r] = 0.f; }
#pragma unroll
  for (int j = 0; j < 4; ++j) O[j] = zero8();

  int n = lane & 15, mAdd = (lane >> 4) * 8;
  int kEnd = q0 + 16;
  for (int k0 = 0; k0 < kEnd; k0 += 32) {
    // S[q, k] = Qeff . K, two 16-col subtiles, d accumulated in two 32-chunks
    v8f s0 = zero8(), s1 = zero8();
    {
      v16bf b0 = load_b(Kp, k0, DHd, 0, lane);
      v16bf b1 = load_b(Kp, k0, DHd, 32, lane);
      s0 = wmma_bf(a0, b0, s0);
      s0 = wmma_bf(a1, b1, s0);
      v16bf b2 = load_b(Kp, k0 + 16, DHd, 0, lane);
      v16bf b3 = load_b(Kp, k0 + 16, DHd, 32, lane);
      s1 = wmma_bf(a0, b2, s1);
      s1 = wmma_bf(a1, b3, s1);
    }
#pragma unroll
    for (int r = 0; r < 8; ++r) {
      int qrow = q0 + mAdd + r;
      float v0 = s0[r] * 0.125f + (((k0 + n) > qrow) ? -1e9f : 0.f);
      float v1 = s1[r] * 0.125f + (((k0 + 16 + n) > qrow) ? -1e9f : 0.f);
      float mx = fmaxf(v0, v1);
#pragma unroll
      for (int off = 1; off < 16; off <<= 1) mx = fmaxf(mx, __shfl_xor(mx, off));
      float mnew = fmaxf(mrun[r], mx);
      float sc = __expf(mrun[r] - mnew);
      float p0 = __expf(v0 - mnew), p1 = __expf(v1 - mnew);
      float rs = p0 + p1;
#pragma unroll
      for (int off = 1; off < 16; off <<= 1) rs += __shfl_xor(rs, off);
      lrun[r] = lrun[r] * sc + rs;
      mrun[r] = mnew;
#pragma unroll
      for (int j = 0; j < 4; ++j) O[j][r] *= sc;
      int prow = mAdd + r;
      lds[wvid][prow * 32 + n] = f2bf(p0);
      lds[wvid][prow * 32 + 16 + n] = f2bf(p1);
    }
    // P (16x32) back as A-fragment from LDS; O += P @ V  (V transposed -> contiguous B frags)
    v16bf pf = load_a(&lds[wvid][0], 0, 32, 0, lane);
#pragma unroll
    for (int j = 0; j < 4; ++j) {
      v16bf bv = load_b(Vp, j * 16, Lseq, k0, lane);
      O[j] = wmma_bf(pf, bv, O[j]);
    }
  }
  int b = bh >> 4, h = bh & 15;
#pragma unroll
  for (int j = 0; j < 4; ++j)
#pragma unroll
    for (int r = 0; r < 8; ++r) {
      float inv = 1.f / lrun[r];
      int row = b * Lseq + q0 + mAdd + r;
      int col = h * DHd + j * 16 + n;
      Abf[(size_t)row * Dm + col] = f2bf(O[j][r] * inv);
    }
}

// ---------------- output GEMM: out = attn @ Wo^T (fp32 out) ----------------
__global__ void __launch_bounds__(256) gemm_out_kernel(const bf16_t* __restrict__ A,
    const bf16_t* __restrict__ W, float* __restrict__ C) {
  int lane = threadIdx.x & 31;
  int t = blockIdx.x * 8 + (threadIdx.x >> 5);
  int m0 = (t >> 4) * 16, n0 = (t & 15) * 64;
  v8f acc[4];
#pragma unroll
  for (int j = 0; j < 4; ++j) acc[j] = zero8();
  for (int k0 = 0; k0 < Dm; k0 += 32) {
    v16bf a = load_a(A, m0, Dm, k0, lane);
#pragma unroll
    for (int j = 0; j < 4; ++j) {
      v16bf b = load_b(W, n0 + j * 16, Dm, k0, lane);
      acc[j] = wmma_bf(a, b, acc[j]);
    }
  }
  int n = lane & 15, mAdd = (lane >> 4) * 8;
#pragma unroll
  for (int j = 0; j < 4; ++j)
#pragma unroll
    for (int r = 0; r < 8; ++r)
      C[(size_t)(m0 + mAdd + r) * Dm + n0 + j * 16 + n] = acc[j][r];
}

extern "C" void kernel_launch(void* const* d_in, const int* in_sizes, int n_in,
                              void* d_out, int out_size, void* d_ws, size_t ws_size,
                              hipStream_t stream) {
  (void)in_sizes; (void)n_in; (void)out_size; (void)ws_size;
  const float* x  = (const float*)d_in[0];
  const float* Wq = (const float*)d_in[1];
  const float* Wk = (const float*)d_in[2];
  const float* Wv = (const float*)d_in[3];
  const float* Wo = (const float*)d_in[4];
  const unsigned char* tmask = (const unsigned char*)d_in[5];
  // d_in[6] (attn_mask) is the pure causal -1e9 mask; applied analytically in flash_kernel.

  char* w = (char*)d_ws;
  auto take = [&](size_t bytes) {
    char* p = w;
    w += (bytes + 255) & ~(size_t)255;
    return p;
  };
  bf16_t* xbf = (bf16_t*)take((size_t)MLrows * Dm * 2);
  bf16_t* wqb = (bf16_t*)take((size_t)Dm * Dm * 2);
  bf16_t* wkb = (bf16_t*)take((size_t)Dm * Dm * 2);
  bf16_t* wvb = (bf16_t*)take((size_t)Dm * Dm * 2);
  bf16_t* wob = (bf16_t*)take((size_t)Dm * Dm * 2);
  float*  Qf  = (float*) take((size_t)MLrows * Dm * 4);
  bf16_t* Qe  = (bf16_t*)take((size_t)MLrows * Dm * 2);
  bf16_t* Kb  = (bf16_t*)take((size_t)MLrows * Dm * 2);
  bf16_t* Vt  = (bf16_t*)take((size_t)MLrows * Dm * 2);
  bf16_t* Abf = (bf16_t*)take((size_t)MLrows * Dm * 2);

  int nx = MLrows * Dm, nw = Dm * Dm;
  cvt_kernel<<<(nx + 255) / 256, 256, 0, stream>>>(x, xbf, nx);
  cvt_kernel<<<(nw + 255) / 256, 256, 0, stream>>>(Wq, wqb, nw);
  cvt_kernel<<<(nw + 255) / 256, 256, 0, stream>>>(Wk, wkb, nw);
  cvt_kernel<<<(nw + 255) / 256, 256, 0, stream>>>(Wv, wvb, nw);
  cvt_kernel<<<(nw + 255) / 256, 256, 0, stream>>>(Wo, wob, nw);

  gemm_qkv_kernel<<<256, 256, 0, stream>>>(xbf, wqb, wkb, wvb, Qf, Kb, Vt);
  lorentz_kernel<<<4096, 256, 0, stream>>>(Qf, tmask, Qe);
  flash_kernel<<<256, 256, 0, stream>>>(Qe, Kb, Vt, Abf);
  gemm_out_kernel<<<256, 256, 0, stream>>>(Abf, wob, (float*)d_out);
}
